// SequentialCNNPoseLSTM_48799418417253
// MI455X (gfx1250) — matrix operand order repaired and checked by reading, not compiled, not run on hardware
//
#include <hip/hip_runtime.h>
#include <hip/hip_bf16.h>
#include <math.h>

// Problem constants (match reference)
#define B_    64
#define CIN   16
#define TT    2048
#define F_    64
#define HID_  256
#define G4    1024   // 4*HID
#define OUTC  20
#define EPSV  1e-5f

// Global (device memory) address space, so laundered pointers still lower to
// global_load_* (LOADcnt only) instead of flat_load_* (LOADcnt+DScnt).
#define GAS __attribute__((address_space(1)))

typedef __attribute__((ext_vector_type(16))) __bf16 v16bf;
typedef __attribute__((ext_vector_type(8)))  __bf16 v8bf;
typedef __attribute__((ext_vector_type(8)))  float  v8f;

union bf16x16 {
  v16bf v;
  __bf16 s[16];
};

// ---------------------------------------------------------------------------
// Conv1d(k=3,pad=1) + bias + ReLU, with deterministic per-block partial sums
// for BatchNorm batch statistics. y layout: (B, F, T). One block = one
// (t-chunk, f, b) triple, 256 threads over t.
// ---------------------------------------------------------------------------
__global__ __launch_bounds__(256) void conv_relu_kernel(
    const float* __restrict__ x, const float* __restrict__ w,
    const float* __restrict__ bias, float* __restrict__ y,
    float* __restrict__ psum, float* __restrict__ psumsq)
{
  const int tc = blockIdx.x, f = blockIdx.y, b = blockIdx.z;
  const int tid = threadIdx.x;
  const int t = tc * 256 + tid;
  __shared__ float wsh[CIN * 3];
  __shared__ float rs[256];
  __shared__ float rq[256];
  if (tid < CIN * 3) wsh[tid] = w[f * CIN * 3 + tid];
  __syncthreads();
  float acc = bias[f];
  for (int c = 0; c < CIN; ++c) {
    const float* xp = x + ((size_t)b * CIN + c) * TT;
#pragma unroll
    for (int k = 0; k < 3; ++k) {
      int ts = t + k - 1;
      float xv = (ts >= 0 && ts < TT) ? xp[ts] : 0.f;
      acc += xv * wsh[c * 3 + k];
    }
  }
  acc = fmaxf(acc, 0.f);
  y[((size_t)b * F_ + f) * TT + t] = acc;
  rs[tid] = acc;
  rq[tid] = acc * acc;
  __syncthreads();
  for (int s = 128; s > 0; s >>= 1) {
    if (tid < s) { rs[tid] += rs[tid + s]; rq[tid] += rq[tid + s]; }
    __syncthreads();
  }
  if (tid == 0) {
    int p = (f * B_ + b) * (TT / 256) + tc;
    psum[p] = rs[0];
    psumsq[p] = rq[0];
  }
}

// ---------------------------------------------------------------------------
// Finalize BN statistics: per-channel scale/shift from partial sums.
// ---------------------------------------------------------------------------
__global__ void bn_stats_kernel(
    const float* __restrict__ psum, const float* __restrict__ psumsq,
    const float* __restrict__ gamma, const float* __restrict__ beta,
    float* __restrict__ scale, float* __restrict__ shift)
{
  int f = threadIdx.x;
  if (f >= F_) return;
  const int NP = B_ * (TT / 256);
  float s = 0.f, q = 0.f;
  for (int i = 0; i < NP; ++i) {
    s += psum[f * NP + i];
    q += psumsq[f * NP + i];
  }
  const float inv = 1.f / (float)(B_ * TT);
  float mean = s * inv;
  float var  = q * inv - mean * mean;
  float sc = gamma[f] * rsqrtf(var + EPSV);
  scale[f] = sc;
  shift[f] = beta[f] - mean * sc;
}

// ---------------------------------------------------------------------------
// Apply BN and transpose (B,F,T) -> (T,B,F) bf16 features for the LSTM.
// ---------------------------------------------------------------------------
__global__ __launch_bounds__(256) void bn_apply_kernel(
    const float* __restrict__ y, const float* __restrict__ scale,
    const float* __restrict__ shift, __bf16* __restrict__ feats)
{
  size_t idx = (size_t)blockIdx.x * 256 + threadIdx.x;   // (t,b,f), f fastest
  int f = (int)(idx & 63);
  int b = (int)((idx >> 6) & 63);
  int t = (int)(idx >> 12);
  float v = y[((size_t)b * F_ + f) * TT + t];
  feats[idx] = (__bf16)(v * scale[f] + shift[f]);
}

// ---------------------------------------------------------------------------
// Pack [w_ih | w_hh] (each (G4, K) f32 row-major, K-dim concatenated) into the
// WMMA bf16 B-fragment layout:
//   fragment (kc, nt): 32 lanes x 16 halves; lane L -> N = nt*16 + (L&15),
//   K = kc*32 + 16*(L>>4) + e  (e = element 0..15, contiguous in memory).
// One thread per (fragment, lane).
// ---------------------------------------------------------------------------
__global__ __launch_bounds__(256) void pack_weights_kernel(
    const float* __restrict__ w_ih, const float* __restrict__ w_hh,
    __bf16* __restrict__ dst, int kin)
{
  int gid = blockIdx.x * 256 + threadIdx.x;
  int lane = gid & 31;
  int frag = gid >> 5;
  int kc = frag >> 6;        // 64 column tiles (G4/16)
  int nt = frag & 63;
  int n  = nt * 16 + (lane & 15);
  int kb = kc * 32 + (lane >> 4) * 16;
  __bf16* d = dst + (size_t)frag * 512 + lane * 16;
#pragma unroll
  for (int e = 0; e < 16; ++e) {
    int K = kb + e;
    float v = (K < kin) ? w_ih[(size_t)n * kin + K]
                        : w_hh[(size_t)n * HID_ + (K - kin)];
    d[e] = (__bf16)v;
  }
}

__global__ void bias_combine_kernel(const float* __restrict__ a,
                                    const float* __restrict__ b,
                                    float* __restrict__ dst)
{
  int i = blockIdx.x * 256 + threadIdx.x;
  dst[i] = a[i] + b[i];
}

__device__ __forceinline__ float sigmoidf_(float v) {
  return 1.f / (1.f + expf(-v));
}

// ---------------------------------------------------------------------------
// Persistent single-workgroup LSTM layer. 1024 threads = 32 waves on one WGP
// (8 waves per SIMD32 for latency hiding). Per timestep:
//   gates(64x1024) = [x_t | h] (64 x KTOT bf16, LDS) @ Wpk (KTOT x 1024 bf16,
//   packed, L2-resident) via v_wmma_f32_16x16x32_bf16.
// Activations -> bf16 gate buffer in LDS; cell state c stays in VGPRs (16 per
// thread); h written back into the LDS A-buffer and streamed to global bf16.
// Wave w: row-tile mt = w>>3 (16 rows of B), col-tiles (w&7)*8 .. +8.
// The packed-weight base pointer is laundered through an empty asm each
// timestep (so LICM cannot hoist-and-spill the t-invariant weight loads) but
// is rebuilt as an address_space(1) pointer so the loads stay global_load_*.
// ---------------------------------------------------------------------------
template <int KIN>
__global__ __launch_bounds__(1024, 1) void lstm_layer_kernel(
    const __bf16* __restrict__ xin,   // (T, B, KIN) bf16
    const __bf16* __restrict__ wpk,   // packed (KC, 64, 32, 16) bf16
    const float*  __restrict__ bias,  // (1024) f32, b_ih + b_hh
    __bf16* __restrict__ hout)        // (T, B, HID) bf16
{
  constexpr int KTOT = KIN + HID_;
  constexpr int KC   = KTOT / 32;
  constexpr int AST  = KTOT + 8;    // row stride (multiple of 8 -> 16B rows)
  constexpr int GST  = G4 + 8;
  constexpr int XCH  = (B_ * KIN) / 8;   // x_t staging chunks of 8 bf16
  __shared__ __align__(16) __bf16 Abuf[B_ * AST];   // [x_t | h]
  __shared__ __align__(16) __bf16 Gbuf[B_ * GST];   // activated gates

  const int tid   = threadIdx.x;
  const int lane  = tid & 31;
  const int wave  = tid >> 5;
  const int mt    = wave >> 3;
  const int ntb   = (wave & 7) * 8;
  const int m0    = mt * 16;
  const int laneM = lane & 15;
  const int laneH = lane >> 4;

  // Zero A-buffer (h starts at 0; x region overwritten every step).
  for (int i = tid; i < B_ * AST; i += 1024) Abuf[i] = (__bf16)0.f;

  // Cell state: thread owns b = tid>>4, j in [ (tid&15)*16, +16 ).
  float c[16];
#pragma unroll
  for (int e = 0; e < 16; ++e) c[e] = 0.f;
  const int cb = tid >> 4;
  const int j0 = (tid & 15) * 16;

  // Per-thread invariants for the activation phase.
  float biasr[8];
  int   gater[8];
  int   colr[8];
#pragma unroll
  for (int nt = 0; nt < 8; ++nt) {
    int colg = (ntb + nt) * 16 + laneM;
    biasr[nt] = bias[colg];
    gater[nt] = colg >> 8;     // 0,1,3 = sigmoid; 2 = tanh (i,f,g,o order)
    colr[nt]  = colg;
  }

  const int arow = (m0 + laneM) * AST;    // A-fragment LDS row base

  // Opaque copy of the per-lane packed-weight base. Re-laundered every
  // timestep inside the loop so weight loads stay in the loop body.
  unsigned long long wbase = (unsigned long long)(wpk + lane * 16);

  __syncthreads();

  for (int t = 0; t < TT; ++t) {
    // ---- stage x_t into A-buffer columns [0, KIN) -------------------------
    const __bf16* xt = xin + (size_t)t * B_ * KIN;
    for (int i = tid; i < XCH; i += 1024) {
      const int el = i * 8;                  // multiple of 8 -> 16B aligned
      const int b  = el / KIN;
      const int k  = el - b * KIN;
      *(v8bf*)&Abuf[b * AST + k] = *(const v8bf*)&xt[el];
      if (t + 1 < TT)                        // warm L2/WGP$ for next step
        __builtin_prefetch(xt + B_ * KIN + el, 0, 3);
    }
    __syncthreads();

    // ---- GEMM: gates = [x|h] @ Wpk ---------------------------------------
    asm volatile("" : "+v"(wbase));          // defeat LICM of weight loads
    const GAS __bf16* wl = (const GAS __bf16*)wbase;   // global AS -> global_load

    v8f acc[8];
#pragma unroll
    for (int nt = 0; nt < 8; ++nt)
#pragma unroll
      for (int r = 0; r < 8; ++r) acc[nt][r] = 0.f;

#pragma unroll 1
    for (int kc = 0; kc < KC; ++kc) {
      // A fragment from LDS (contiguous k order 16*laneH + s, as in packed B).
      const int kb = kc * 32 + 16 * laneH;
      v8bf lo = *(const v8bf*)&Abuf[arow + kb];
      v8bf hi = *(const v8bf*)&Abuf[arow + kb + 8];
      v16bf a = __builtin_shufflevector(lo, hi,
          0, 1, 2, 3, 4, 5, 6, 7, 8, 9, 10, 11, 12, 13, 14, 15);
      // Issue all 8 B-fragment loads, then the WMMA chain.
      const GAS __bf16* wk = wl + (size_t)(kc * 64 + ntb) * 512;
      v16bf bfr[8];
#pragma unroll
      for (int nt = 0; nt < 8; ++nt)
        bfr[nt] = *(const GAS v16bf*)(wk + nt * 512);
#pragma unroll
      for (int nt = 0; nt < 8; ++nt)
        acc[nt] = __builtin_amdgcn_wmma_f32_16x16x32_bf16(
            false, a, false, bfr[nt], (short)0, acc[nt], false, false);
    }

    // ---- activations -> bf16 gate buffer ---------------------------------
#pragma unroll
    for (int nt = 0; nt < 8; ++nt) {
      const int colg = colr[nt];
      const bool is_g = (gater[nt] == 2);
#pragma unroll
      for (int r = 0; r < 8; ++r) {
        const int bb = m0 + laneH * 8 + r;   // C-layout: lane half selects M+8
        float v = acc[nt][r] + biasr[nt];
        v = is_g ? tanhf(v) : sigmoidf_(v);
        Gbuf[bb * GST + colg] = (__bf16)v;
      }
    }
    __syncthreads();

    // ---- cell update: c = f*c + i*g ; h = o*tanh(c) ----------------------
    bf16x16 gi, gf, gg, go, hv;
    gi.v = *(const v16bf*)&Gbuf[cb * GST + j0];
    gf.v = *(const v16bf*)&Gbuf[cb * GST + 256 + j0];
    gg.v = *(const v16bf*)&Gbuf[cb * GST + 512 + j0];
    go.v = *(const v16bf*)&Gbuf[cb * GST + 768 + j0];
#pragma unroll
    for (int e = 0; e < 16; ++e) {
      c[e] = (float)gf.s[e] * c[e] + (float)gi.s[e] * (float)gg.s[e];
      hv.s[e] = (__bf16)((float)go.s[e] * tanhf(c[e]));
    }
    *(v16bf*)&Abuf[cb * AST + KIN + j0] = hv.v;
    *(v16bf*)(hout + (size_t)t * B_ * HID_ + (size_t)cb * HID_ + j0) = hv.v;
    __syncthreads();
  }
}

// ---------------------------------------------------------------------------
// FC per timestep: out(B,OUT,T) = h2(T,B,H) @ fc_w^T + fc_b. 20 outputs ->
// VALU with weights staged in LDS (broadcast reads).
// ---------------------------------------------------------------------------
__global__ __launch_bounds__(256) void fc_kernel(
    const __bf16* __restrict__ h2, const float* __restrict__ fw,
    const float* __restrict__ fb, float* __restrict__ out)
{
  __shared__ float wsm[OUTC * HID_];
  for (int i = threadIdx.x; i < OUTC * HID_; i += 256) wsm[i] = fw[i];
  __syncthreads();
  int pair = blockIdx.x * 256 + threadIdx.x;   // pair = t*B + b
  int b = pair & 63;
  int t = pair >> 6;
  const __bf16* hr = h2 + (size_t)pair * HID_;
  float acc[OUTC];
#pragma unroll
  for (int o = 0; o < OUTC; ++o) acc[o] = fb[o];
  for (int k = 0; k < HID_; ++k) {
    float h = (float)hr[k];
#pragma unroll
    for (int o = 0; o < OUTC; ++o) acc[o] += h * wsm[o * HID_ + k];
  }
#pragma unroll
  for (int o = 0; o < OUTC; ++o)
    out[((size_t)b * OUTC + o) * TT + t] = acc[o];
}

// ---------------------------------------------------------------------------
extern "C" void kernel_launch(void* const* d_in, const int* in_sizes, int n_in,
                              void* d_out, int out_size, void* d_ws, size_t ws_size,
                              hipStream_t stream)
{
  const float* x      = (const float*)d_in[0];
  const float* conv_w = (const float*)d_in[1];
  const float* conv_b = (const float*)d_in[2];
  const float* gamma  = (const float*)d_in[3];
  const float* beta   = (const float*)d_in[4];
  const float* wih0   = (const float*)d_in[5];
  const float* whh0   = (const float*)d_in[6];
  const float* bih0   = (const float*)d_in[7];
  const float* bhh0   = (const float*)d_in[8];
  const float* wih1   = (const float*)d_in[9];
  const float* whh1   = (const float*)d_in[10];
  const float* bih1   = (const float*)d_in[11];
  const float* bhh1   = (const float*)d_in[12];
  const float* fcw    = (const float*)d_in[13];
  const float* fcb    = (const float*)d_in[14];
  float* out = (float*)d_out;

  // Workspace carve-up (aligned to 256B).
  char* p = (char*)d_ws;
  auto alloc = [&](size_t bytes) -> char* {
    char* r = p;
    p += (bytes + 255) & ~(size_t)255;
    return r;
  };
  float*  y      = (float*) alloc(sizeof(float)  * (size_t)B_ * F_ * TT);
  __bf16* feats  = (__bf16*)alloc(sizeof(__bf16) * (size_t)TT * B_ * F_);
  __bf16* h1     = (__bf16*)alloc(sizeof(__bf16) * (size_t)TT * B_ * HID_);
  __bf16* h2     = (__bf16*)alloc(sizeof(__bf16) * (size_t)TT * B_ * HID_);
  __bf16* wpk0   = (__bf16*)alloc(sizeof(__bf16) * ((F_ + HID_) / 32) * 64 * 512);
  __bf16* wpk1   = (__bf16*)alloc(sizeof(__bf16) * ((HID_ + HID_) / 32) * 64 * 512);
  float*  bias0  = (float*) alloc(sizeof(float) * G4);
  float*  bias1  = (float*) alloc(sizeof(float) * G4);
  float*  scale  = (float*) alloc(sizeof(float) * F_);
  float*  shift  = (float*) alloc(sizeof(float) * F_);
  float*  psum   = (float*) alloc(sizeof(float) * F_ * B_ * (TT / 256));
  float*  psumsq = (float*) alloc(sizeof(float) * F_ * B_ * (TT / 256));
  (void)ws_size; (void)in_sizes; (void)n_in; (void)out_size;

  // 1) Conv + ReLU + BN partial stats
  conv_relu_kernel<<<dim3(TT / 256, F_, B_), 256, 0, stream>>>(
      x, conv_w, conv_b, y, psum, psumsq);
  // 2) BN scale/shift
  bn_stats_kernel<<<1, 64, 0, stream>>>(psum, psumsq, gamma, beta, scale, shift);
  // 3) BN apply + transpose to (T,B,F) bf16
  bn_apply_kernel<<<(TT * B_ * F_) / 256, 256, 0, stream>>>(y, scale, shift, feats);
  // 4) Pack weights into WMMA fragment layout + combine biases
  {
    int frags0 = ((F_ + HID_) / 32) * 64;     // 640
    int frags1 = ((HID_ + HID_) / 32) * 64;   // 1024
    pack_weights_kernel<<<frags0 * 32 / 256, 256, 0, stream>>>(wih0, whh0, wpk0, F_);
    pack_weights_kernel<<<frags1 * 32 / 256, 256, 0, stream>>>(wih1, whh1, wpk1, HID_);
    bias_combine_kernel<<<G4 / 256, 256, 0, stream>>>(bih0, bhh0, bias0);
    bias_combine_kernel<<<G4 / 256, 256, 0, stream>>>(bih1, bhh1, bias1);
  }
  // 5) Persistent LSTM layers (one WGP each, sequential over T)
  lstm_layer_kernel<F_  ><<<1, 1024, 0, stream>>>(feats, wpk0, bias0, h1);
  lstm_layer_kernel<HID_><<<1, 1024, 0, stream>>>(h1,    wpk1, bias1, h2);
  // 6) FC per timestep -> (B, OUT, T)
  fc_kernel<<<(TT * B_) / 256, 256, 0, stream>>>(h2, fcw, fcb, out);
}
